// TransformerBlock_surface_75161927680021
// MI455X (gfx1250) — matrix operand order repaired and checked by reading, not compiled, not run on hardware
//
#include <hip/hip_runtime.h>
#include <hip/hip_bf16.h>

// ---------------------------------------------------------------------------
// Problem constants (reference: B=8, N=2048, M=2048, DP=128, DM=256, K=16)
// ---------------------------------------------------------------------------
#define BB   8
#define NN   2048
#define MM   2048
#define DP   128
#define DM   256
#define KNN  16
#define MNROWS (BB * NN)   // 16384

// ---------------------------------------------------------------------------
// Types for CDNA5 WMMA
// ---------------------------------------------------------------------------
typedef __bf16 bf16x16 __attribute__((ext_vector_type(16)));
typedef float  f32x8   __attribute__((ext_vector_type(8)));

struct alignas(16) V4 { unsigned int x, y, z, w; };
struct V8 { V4 lo, hi; };   // 32 bytes == bf16x16

__device__ __forceinline__ unsigned short f2bf(float x) {
    unsigned int u = __float_as_uint(x);
    unsigned int r = (u + 0x7FFFu + ((u >> 16) & 1u)) >> 16;
    return (unsigned short)r;
}
__device__ __forceinline__ float bf2f(unsigned short h) {
    return __uint_as_float(((unsigned int)h) << 16);
}

__device__ __forceinline__ bf16x16 load_frag_split(const unsigned short* p) {
    // A-matrix per-lane layout: K runs [0..7] and [16..23] (two 16B runs, 16-short gap)
    V8 v;
    v.lo = *(const V4*)(p);
    v.hi = *(const V4*)(p + 16);
    return __builtin_bit_cast(bf16x16, v);
}
__device__ __forceinline__ bf16x16 load_frag_contig(const unsigned short* p) {
    // per-lane layout used here for both A and B: 16 contiguous K values (32B run)
    V8 v;
    v.lo = *(const V4*)(p);
    v.hi = *(const V4*)(p + 8);
    return __builtin_bit_cast(bf16x16, v);
}

// ---------------------------------------------------------------------------
// fp32 -> bf16 conversion (vectorized x4; all sizes are multiples of 1024)
// ---------------------------------------------------------------------------
__global__ void cvt_bf16_kernel(const float* __restrict__ in,
                                unsigned short* __restrict__ out, int n4) {
    int i = blockIdx.x * blockDim.x + threadIdx.x;
    if (i < n4) {
        const float* p = in + i * 4;
        ushort4 o;
        o.x = f2bf(p[0]); o.y = f2bf(p[1]); o.z = f2bf(p[2]); o.w = f2bf(p[3]);
        *(ushort4*)(out + i * 4) = o;
    }
}

// ---------------------------------------------------------------------------
// fp32 [K,256] -> bf16 [256,K] tiled transpose (weights; coalesced both sides)
// grid = (K/32, 256/32), block = (32, 8)
// ---------------------------------------------------------------------------
__global__ void cvt_transpose_kernel(const float* __restrict__ in,
                                     unsigned short* __restrict__ out, int K) {
    __shared__ float t[32][33];
    const int k0 = blockIdx.x * 32, n0 = blockIdx.y * 32;
    for (int i = threadIdx.y; i < 32; i += 8)
        t[i][threadIdx.x] = in[(size_t)(k0 + i) * DM + n0 + threadIdx.x];
    __syncthreads();
    for (int i = threadIdx.y; i < 32; i += 8)
        out[(size_t)(n0 + i) * K + k0 + threadIdx.x] = f2bf(t[threadIdx.x][i]);
}

// ---------------------------------------------------------------------------
// KNN: per thread one query point, register-resident sorted top-16,
// surface points tiled through LDS.  grid = B*(N/256), block = 256
// ---------------------------------------------------------------------------
__global__ __launch_bounds__(256)
void knn_kernel(const float* __restrict__ xyz, const float* __restrict__ xyzS,
                int* __restrict__ knn) {
    __shared__ float sx[256 * 3];
    const int tid = threadIdx.x;
    const int bb  = blockIdx.x >> 3;
    const int n   = ((blockIdx.x & 7) << 8) + tid;

    const float* xa = xyz + ((size_t)(bb * NN + n)) * 3;
    const float qx = xa[0], qy = xa[1], qz = xa[2];

    float bd[KNN];
    int   bi[KNN];
#pragma unroll
    for (int s = 0; s < KNN; ++s) { bd[s] = 1e30f; bi[s] = 0; }

    for (int t0 = 0; t0 < MM; t0 += 256) {
        __syncthreads();
#pragma unroll
        for (int it = 0; it < 3; ++it) {
            int i = it * 256 + tid;
            sx[i] = xyzS[((size_t)bb * MM + t0) * 3 + i];
        }
        __syncthreads();
        for (int j = 0; j < 256; ++j) {
            float dx = qx - sx[j * 3 + 0];
            float dy = qy - sx[j * 3 + 1];
            float dz = qz - sx[j * 3 + 2];
            float d  = fmaf(dx, dx, fmaf(dy, dy, dz * dz));
            if (d < bd[KNN - 1]) {
                bd[KNN - 1] = d; bi[KNN - 1] = t0 + j;
#pragma unroll
                for (int s = KNN - 1; s > 0; --s) {
                    if (bd[s] < bd[s - 1]) {
                        float td = bd[s]; bd[s] = bd[s - 1]; bd[s - 1] = td;
                        int   ti = bi[s]; bi[s] = bi[s - 1]; bi[s - 1] = ti;
                    }
                }
            }
        }
    }
#pragma unroll
    for (int s = 0; s < KNN; ++s)
        knn[((size_t)(bb * NN + n)) * KNN + s] = bi[s];
}

// ---------------------------------------------------------------------------
// Generic bf16 WMMA GEMM: C[M,256] = A[M,KD] @ B[KD,256] (+bias).
// B is supplied PRE-TRANSPOSED as BT[256][KD] so both stagings are b128 copies.
// Workgroup tile 128x64, 8 waves (wave w -> row-tile w, 4 col tiles).
// grid = (M/128, 4), block = 256.  KD is compile-time (128 or 256).
// ---------------------------------------------------------------------------
template <int KD, bool HASBIAS, bool OUTF, bool OUTB>
__global__ __launch_bounds__(256)
void gemm_bf16_kernel(const unsigned short* __restrict__ Ag,
                      const unsigned short* __restrict__ BTg,
                      const float* __restrict__ bias,
                      float* __restrict__ Cf,
                      unsigned short* __restrict__ Cb) {
    __shared__ unsigned short As[128 * 40];   // stride 40 shorts (16B-aligned runs)
    __shared__ unsigned short Bs[64 * 40];    // [col][k]

    const int tid  = threadIdx.x;
    const int w    = tid >> 5;
    const int lane = tid & 31;
    const int m16  = lane & 15;
    const int hig  = lane >> 4;
    const int row0 = blockIdx.x * 128;
    const int col0 = blockIdx.y * 64;

    f32x8 acc[4] = {};

    for (int kb = 0; kb < KD; kb += 32) {
        if (kb + 32 < KD)   // next A tile prefetch (global_prefetch_b8)
            __builtin_prefetch(Ag + (size_t)(row0 + (tid >> 1)) * KD + kb + 32, 0, 3);
        __syncthreads();
        // stage A [128][32]: 512 x b128, 2 per thread
#pragma unroll
        for (int it = 0; it < 2; ++it) {
            int i = it * 256 + tid;
            int r = i >> 2, c = (i & 3) * 8;
            *(V4*)(As + r * 40 + c) =
                *(const V4*)(Ag + (size_t)(row0 + r) * KD + kb + c);
        }
        // stage BT [64][32]: 256 x b128, 1 per thread
        {
            int r = tid >> 2, c = (tid & 3) * 8;
            *(V4*)(Bs + r * 40 + c) =
                *(const V4*)(BTg + (size_t)(col0 + r) * KD + kb + c);
        }
        __syncthreads();

        bf16x16 av = load_frag_contig(As + (w * 16 + m16) * 40 + hig * 16);
        bf16x16 bv[4];
#pragma unroll
        for (int ct = 0; ct < 4; ++ct)
            bv[ct] = load_frag_contig(Bs + (ct * 16 + m16) * 40 + hig * 16);
#pragma unroll
        for (int ct = 0; ct < 4; ++ct)
            acc[ct] = __builtin_amdgcn_wmma_f32_16x16x32_bf16(
                false, av, false, bv[ct], (short)0, acc[ct], false, false);
    }

#pragma unroll
    for (int ct = 0; ct < 4; ++ct) {
        int col = col0 + ct * 16 + m16;
        float bs = HASBIAS ? bias[col] : 0.0f;
#pragma unroll
        for (int r = 0; r < 8; ++r) {
            int row = row0 + w * 16 + r + 8 * hig;
            float v = acc[ct][r] + bs;
            if (OUTF) Cf[(size_t)row * DM + col] = v;
            if (OUTB) Cb[(size_t)row * DM + col] = f2bf(v);
        }
    }
}

// ---------------------------------------------------------------------------
// Workgroup-level GEMM inside the fused kernel:
//   out[64,256] = A_lds[64,256](bf16) @ W (bf16, pre-transposed [256][256]) + bias
// 8 waves: wave w -> row-tile (w&3), col-half (w>>2).
// Barrier discipline allows in-place A overwrite in the epilogue.
// ---------------------------------------------------------------------------
#define SA 264   // LDS row stride for 64x256 bf16 tiles

template <bool RELU, bool OUTB>
__device__ __forceinline__ void wg_gemm(const unsigned short* A,
                                        const unsigned short* __restrict__ BTg,
                                        const float* __restrict__ bias,
                                        unsigned short* Bt,
                                        float* outF, unsigned short* outB,
                                        int tid) {
    const int w    = tid >> 5;
    const int lane = tid & 31;
    const int rt   = w & 3;
    const int ch   = w >> 2;
    const int m16  = lane & 15;
    const int hig  = lane >> 4;

    f32x8 acc[8] = {};

    for (int kb = 0; kb < DM; kb += 32) {
        if (kb + 32 < DM)
            __builtin_prefetch(BTg + (size_t)tid * DM + kb + 32, 0, 3);
        __syncthreads();                       // Bt reuse + previous epilogue done
        // stage BT slice [256 cols][32 k]: 1024 x b128, 4 per thread
#pragma unroll
        for (int it = 0; it < 4; ++it) {
            int i = it * 256 + tid;
            int r = i >> 2, c = (i & 3) * 8;
            *(V4*)(Bt + r * 40 + c) = *(const V4*)(BTg + (size_t)r * DM + kb + c);
        }
        __syncthreads();

        bf16x16 av = load_frag_contig(A + (rt * 16 + m16) * SA + kb + hig * 16);
        bf16x16 bv[8];
#pragma unroll
        for (int c8 = 0; c8 < 8; ++c8)
            bv[c8] = load_frag_contig(Bt + ((ch * 8 + c8) * 16 + m16) * 40 + hig * 16);
#pragma unroll
        for (int c8 = 0; c8 < 8; ++c8)
            acc[c8] = __builtin_amdgcn_wmma_f32_16x16x32_bf16(
                false, av, false, bv[c8], (short)0, acc[c8], false, false);
    }
    __syncthreads();                           // all A reads done before in-place write

#pragma unroll
    for (int c8 = 0; c8 < 8; ++c8) {
        int col = ch * 128 + c8 * 16 + m16;
        float bs = bias[col];
#pragma unroll
        for (int r = 0; r < 8; ++r) {
            int row = rt * 16 + r + 8 * hig;
            float v = acc[c8][r] + bs;
            if (RELU) v = fmaxf(v, 0.0f);
            if (OUTB) outB[row * SA + col] = f2bf(v);
            else      outF[row * DM + col] = v;
        }
    }
    __syncthreads();
}

// ---------------------------------------------------------------------------
// Fused attention: one workgroup = 4 query points (64 rows of K=16 neighbors).
// LDS-resident pipeline: h1 -> pos -> g -> h2 -> scores -> softmax -> res -> out
// grid = B * (N/4) = 4096, block = 256, dynamic LDS ~188 KB
// ---------------------------------------------------------------------------
__global__ __launch_bounds__(256)
void fused_attn_kernel(const float* __restrict__ q,
                       const float* __restrict__ k_all,
                       const float* __restrict__ v_all,
                       const int* __restrict__ knn,
                       const float* __restrict__ xyz,
                       const float* __restrict__ xyzS,
                       const float* __restrict__ features,
                       const float* __restrict__ fd1_w, const float* __restrict__ fd1_b,
                       const unsigned short* __restrict__ fd2T, const float* __restrict__ fd2_b,
                       const unsigned short* __restrict__ fg1T, const float* __restrict__ fg1_b,
                       const unsigned short* __restrict__ fg2T, const float* __restrict__ fg2_b,
                       const float* __restrict__ fc2_w, const float* __restrict__ fc2_b,
                       float* __restrict__ out, float* __restrict__ attn_out) {
    extern __shared__ unsigned char smem[];
    unsigned short* Ah     = (unsigned short*)smem;        // [64][SA] (h1 / g / h2)
    unsigned short* posS   = Ah + 64 * SA;                 // [64][SA]
    unsigned short* vgS    = posS + 64 * SA;               // [64][SA]
    unsigned short* Bt     = vgS + 64 * SA;                // [256*40] weight staging
    float*          scoreS = (float*)(Bt + 256 * 40);      // [64][256]
    float*          qS     = scoreS + 64 * DM;             // [4][256] (reused as res)
    float*          dS     = qS + 4 * DM;                  // [64][4] delta
    int*            idxS   = (int*)(dS + 64 * 4);          // [64]

    const int tid = threadIdx.x;
    const int wg  = blockIdx.x;
    const int bb  = wg >> 9;                 // 512 tiles per batch
    const int n0  = (wg & 511) * 4;

    // --- neighbor indices + deltas + q rows -------------------------------
    if (tid < 64) {
        int r = tid, ln = r >> 4, kk = r & 15;
        int j = knn[((size_t)(bb * NN + n0 + ln)) * KNN + kk];
        idxS[r] = j;
        const float* xa = xyz  + ((size_t)(bb * NN + n0 + ln)) * 3;
        const float* xb = xyzS + ((size_t)(bb * MM + j)) * 3;
        dS[r * 4 + 0] = xa[0] - xb[0];
        dS[r * 4 + 1] = xa[1] - xb[1];
        dS[r * 4 + 2] = xa[2] - xb[2];
    }
#pragma unroll
    for (int it = 0; it < 4; ++it)
        qS[it * DM + tid] = q[((size_t)(bb * NN + n0 + it)) * DM + tid];
    __syncthreads();

    // --- h1 = relu(delta @ fd1 + b)  (3->256, VALU) -----------------------
    {
        float w0 = fd1_w[tid], w1 = fd1_w[DM + tid], w2 = fd1_w[2 * DM + tid];
        float b1 = fd1_b[tid];
#pragma unroll
        for (int r = 0; r < 64; ++r) {
            float a = fmaf(dS[r * 4 + 0], w0,
                      fmaf(dS[r * 4 + 1], w1,
                      fmaf(dS[r * 4 + 2], w2, b1)));
            Ah[r * SA + tid] = f2bf(fmaxf(a, 0.0f));
        }
    }

    // --- pos = h1 @ fd2 + b  (WMMA) ---------------------------------------
    wg_gemm<false, true>(Ah, fd2T, fd2_b, Bt, nullptr, posS, tid);

    // --- g = q - kg + pos ; gather vg (batched loads, coalesced) ----------
#pragma unroll
    for (int r0 = 0; r0 < 64; r0 += 16) {
        float kv[16], vv[16];
#pragma unroll
        for (int r = 0; r < 16; ++r) {
            size_t grow = ((size_t)bb * MM + idxS[r0 + r]) * DM + tid;
            kv[r] = k_all[grow];
            vv[r] = v_all[grow];
        }
#pragma unroll
        for (int r = 0; r < 16; ++r) {
            int rr = r0 + r;
            float g = qS[(rr >> 4) * DM + tid] - kv[r] + bf2f(posS[rr * SA + tid]);
            Ah[rr * SA + tid]  = f2bf(g);
            vgS[rr * SA + tid] = f2bf(vv[r]);
        }
    }

    // --- gamma MLP (WMMA): h2 = relu(g@fg1+b) in-place; scores = h2@fg2+b --
    wg_gemm<true,  true >(Ah, fg1T, fg1_b, Bt, nullptr, Ah, tid);
    wg_gemm<false, false>(Ah, fg2T, fg2_b, Bt, scoreS, nullptr, tid);

    // --- softmax over K (axis=-2) + res = sum_k attn*(vg+pos) -------------
    const float scale = 0.0625f;   // 1/sqrt(DM)
#pragma unroll
    for (int ln = 0; ln < 4; ++ln) {
        float mx = -1e30f;
#pragma unroll
        for (int kk = 0; kk < KNN; ++kk)
            mx = fmaxf(mx, scoreS[(ln * KNN + kk) * DM + tid] * scale);
        float e[KNN], sum = 0.0f;
#pragma unroll
        for (int kk = 0; kk < KNN; ++kk) {
            e[kk] = __expf(scoreS[(ln * KNN + kk) * DM + tid] * scale - mx);
            sum += e[kk];
        }
        float rs = 1.0f / sum;
        float res = 0.0f;
#pragma unroll
        for (int kk = 0; kk < KNN; ++kk) {
            float a = e[kk] * rs;
            int r = ln * KNN + kk;
            attn_out[(((size_t)(bb * NN + n0 + ln)) * KNN + kk) * DM + tid] = a;
            res += a * (bf2f(vgS[r * SA + tid]) + bf2f(posS[r * SA + tid]));
        }
        qS[ln * DM + tid] = res;   // reuse qS as res tile
    }
    __syncthreads();

    // --- out = features + res @ fc2 + b  (256->128, VALU) -----------------
#pragma unroll
    for (int it = 0; it < 2; ++it) {
        int i  = it * 256 + tid;
        int ln = i >> 7, d = i & 127;
        float acc = fc2_b[d] + features[((size_t)(bb * NN + n0 + ln)) * DP + d];
#pragma unroll 8
        for (int f = 0; f < DM; ++f)
            acc = fmaf(qS[ln * DM + f], fc2_w[f * DP + d], acc);
        out[((size_t)(bb * NN + n0 + ln)) * DP + d] = acc;
    }
}

// ---------------------------------------------------------------------------
// Host launcher
// ---------------------------------------------------------------------------
extern "C" void kernel_launch(void* const* d_in, const int* in_sizes, int n_in,
                              void* d_out, int out_size, void* d_ws, size_t ws_size,
                              hipStream_t stream) {
    (void)in_sizes; (void)n_in; (void)out_size; (void)ws_size;

    const float* xyz    = (const float*)d_in[0];
    const float* feat   = (const float*)d_in[1];
    const float* xyzS   = (const float*)d_in[2];
    const float* featS  = (const float*)d_in[3];
    const float* fc1_w  = (const float*)d_in[4];
    const float* fc1_b  = (const float*)d_in[5];
    const float* fc1s_w = (const float*)d_in[6];
    const float* fc1s_b = (const float*)d_in[7];
    const float* fc2_w  = (const float*)d_in[8];
    const float* fc2_b  = (const float*)d_in[9];
    const float* fd1_w  = (const float*)d_in[10];
    const float* fd1_b  = (const float*)d_in[11];
    const float* fd2_w  = (const float*)d_in[12];
    const float* fd2_b  = (const float*)d_in[13];
    const float* fg1_w  = (const float*)d_in[14];
    const float* fg1_b  = (const float*)d_in[15];
    const float* fg2_w  = (const float*)d_in[16];
    const float* fg2_b  = (const float*)d_in[17];
    const float* wk     = (const float*)d_in[18];
    const float* wv     = (const float*)d_in[19];

    unsigned char* ws = (unsigned char*)d_ws;
    size_t off = 0;
    auto alloc = [&](size_t bytes) -> void* {
        void* p = ws + off;
        off = (off + bytes + 255) & ~(size_t)255;
        return p;
    };

    int*            knn    = (int*)alloc((size_t)MNROWS * KNN * 4);
    unsigned short* featB  = (unsigned short*)alloc((size_t)MNROWS * DP * 2);
    unsigned short* featSB = (unsigned short*)alloc((size_t)MNROWS * DP * 2);
    unsigned short* fc1T   = (unsigned short*)alloc((size_t)DP * DM * 2);  // [256][128]
    unsigned short* fc1sT  = (unsigned short*)alloc((size_t)DP * DM * 2);
    unsigned short* wkT    = (unsigned short*)alloc((size_t)DM * DM * 2);  // [256][256]
    unsigned short* wvT    = (unsigned short*)alloc((size_t)DM * DM * 2);
    unsigned short* fd2T   = (unsigned short*)alloc((size_t)DM * DM * 2);
    unsigned short* fg1T   = (unsigned short*)alloc((size_t)DM * DM * 2);
    unsigned short* fg2T   = (unsigned short*)alloc((size_t)DM * DM * 2);
    float*          qF     = (float*)alloc((size_t)MNROWS * DM * 4);
    unsigned short* xsB    = (unsigned short*)alloc((size_t)MNROWS * DM * 2);
    float*          kF     = (float*)alloc((size_t)MNROWS * DM * 4);
    float*          vF     = (float*)alloc((size_t)MNROWS * DM * 4);

    // activations: straight bf16 convert (row-major kept)
    cvt_bf16_kernel<<<(MNROWS * DP / 4 + 255) / 256, 256, 0, stream>>>(
        feat, featB, MNROWS * DP / 4);
    cvt_bf16_kernel<<<(MNROWS * DP / 4 + 255) / 256, 256, 0, stream>>>(
        featS, featSB, MNROWS * DP / 4);

    // weights: convert + transpose to [N][K] for b128 staging
    auto cvtT = [&](const float* src, unsigned short* dst, int K) {
        cvt_transpose_kernel<<<dim3(K / 32, DM / 32), dim3(32, 8), 0, stream>>>(src, dst, K);
    };
    cvtT(fc1_w,  fc1T,  DP);
    cvtT(fc1s_w, fc1sT, DP);
    cvtT(wk,     wkT,   DM);
    cvtT(wv,     wvT,   DM);
    cvtT(fd2_w,  fd2T,  DM);
    cvtT(fg1_w,  fg1T,  DM);
    cvtT(fg2_w,  fg2T,  DM);

    knn_kernel<<<BB * (NN / 256), 256, 0, stream>>>(xyz, xyzS, knn);

    dim3 gg(MNROWS / 128, DM / 64);
    // q = features @ fc1 + b                      -> f32
    gemm_bf16_kernel<DP, true, true, false><<<gg, 256, 0, stream>>>(
        featB, fc1T, fc1_b, qF, nullptr);
    // xs = features_surface @ fc1s + b            -> bf16 (feeds k/v)
    gemm_bf16_kernel<DP, true, false, true><<<gg, 256, 0, stream>>>(
        featSB, fc1sT, fc1s_b, nullptr, xsB);
    // k_all = xs @ wk ; v_all = xs @ wv           -> f32 (gathered later)
    gemm_bf16_kernel<DM, false, true, false><<<gg, 256, 0, stream>>>(
        xsB, wkT, nullptr, kF, nullptr);
    gemm_bf16_kernel<DM, false, true, false><<<gg, 256, 0, stream>>>(
        xsB, wvT, nullptr, vF, nullptr);

    float* outp  = (float*)d_out;
    float* attnp = outp + (size_t)MNROWS * DP;

    size_t lds = (size_t)(3 * 64 * SA + 256 * 40) * 2
               + (size_t)(64 * DM + 4 * DM + 64 * 4) * 4
               + 64 * 4;
    (void)hipFuncSetAttribute((const void*)fused_attn_kernel,
                              hipFuncAttributeMaxDynamicSharedMemorySize, (int)lds);
    fused_attn_kernel<<<BB * (NN / 4), 256, lds, stream>>>(
        qF, kF, vF, knn, xyz, xyzS, feat,
        fd1_w, fd1_b, fd2T, fd2_b, fg1T, fg1_b, fg2T, fg2_b,
        fc2_w, fc2_b, outp, attnp);
}
